// LSTMlayer_48352741818532
// MI455X (gfx1250) — matrix operand order repaired and checked by reading, not compiled, not run on hardware
//
#include <hip/hip_runtime.h>
#include <hip/hip_bf16.h>

// ---------------------------------------------------------------------------
// LSTM layer, T=4096 B=64 I=H=128, fp32 in/out, bf16 WMMA matmuls (f32 acc).
// Phase 1 (wide): P[t,:,:] = W_ih @ x_t^T + b_ih + b_hh   (stored in WMMA
//                 C-fragment layout in workspace)
// Phase 2 (1 persistent WGP, 512 thr / 16 waves): per step
//                 ifgo = W_hh @ h + P[t]; gates fused in registers;
//                 h kept in LDS (bf16), c kept in VGPRs.
//                 16 waves * ~200 VGPRs -> 4 waves/SIMD, no spill pressure.
// ---------------------------------------------------------------------------

typedef __attribute__((ext_vector_type(16))) __bf16 v16bf;
typedef __attribute__((ext_vector_type(8)))  __bf16 v8bf;
typedef __attribute__((ext_vector_type(8)))  float  v8f;
typedef __attribute__((ext_vector_type(4)))  float  v4f;

#define T_TOTAL 4096
#define BATCH   64
#define IDIM    128
#define HDIM    128
#define HSTRIDE 136   // padded bf16 row stride for h in LDS (272B = 16B-aligned)

#if __has_builtin(__builtin_amdgcn_tanhf)
__device__ __forceinline__ float fast_tanh(float x) { return __builtin_amdgcn_tanhf(x); }
#else
__device__ __forceinline__ float fast_tanh(float x) { return tanhf(x); }
#endif
__device__ __forceinline__ float fast_sig(float x) { return 0.5f + 0.5f * fast_tanh(0.5f * x); }

__device__ __forceinline__ v8f wmma_bf16(v16bf a, v16bf b, v8f c) {
    // (neg_a, A, neg_b, B, c_mod, C, reuse_a, reuse_b)
    return __builtin_amdgcn_wmma_f32_16x16x32_bf16(false, a, false, b, (short)0, c,
                                                   false, false);
}

// A-fragment gather (contiguous k order, same as B): lane(=half,ln) holds, for k-chunk kc,
// elements k = kc*32 + 16*half + {0..15}; rows = tile+ln.
__device__ __forceinline__ v16bf load_a_frag_f32(const float* __restrict__ rowbase,
                                                 int kc, int half) {
    const float* p = rowbase + kc * 32 + 16 * half;
    v4f p0 = *(const v4f*)(p);
    v4f p1 = *(const v4f*)(p + 4);
    v4f q0 = *(const v4f*)(p + 8);
    v4f q1 = *(const v4f*)(p + 12);
    v16bf a;
#pragma unroll
    for (int i = 0; i < 4; ++i) {
        a[i]      = (__bf16)p0[i];
        a[4 + i]  = (__bf16)p1[i];
        a[8 + i]  = (__bf16)q0[i];
        a[12 + i] = (__bf16)q1[i];
    }
    return a;
}

// ---------------------------------------------------------------------------
// Phase 1: P^T GEMM.  M = 512 gate rows (32 tiles), N = (t,b), K = 128 (4 chunks).
// One wave per (t, m-tile); each wave does the 4 batch N-tiles of its t.
// Output stored as fragments: P[((t_local*32+mt)*4+bt)*256 + lane*8 + v].
// ---------------------------------------------------------------------------
__global__ __launch_bounds__(256) void lstm_gates_kernel(
    const float* __restrict__ seq, const float* __restrict__ w_ih,
    const float* __restrict__ b_ih, const float* __restrict__ b_hh,
    float* __restrict__ P, int t0) {
    const int tid  = threadIdx.x;
    const int wave = tid >> 5, lane = tid & 31;
    const int half = lane >> 4, ln = lane & 15;
    const int W       = blockIdx.x * 8 + wave;
    const int mt      = W & 31;
    const int t_local = W >> 5;
    const int t       = t0 + t_local;

    // bias per accumulator element: row = mt*16 + v + 8*half (C/D layout)
    float bias[8];
#pragma unroll
    for (int v = 0; v < 8; ++v) {
        int row = mt * 16 + v + 8 * half;
        bias[v] = b_ih[row] + b_hh[row];
    }

    // A fragments: W_ih rows for this m-tile, all 4 k-chunks (bf16, registers)
    v16bf afrag[4];
    const float* arow = w_ih + (size_t)(mt * 16 + ln) * IDIM;
#pragma unroll
    for (int kc = 0; kc < 4; ++kc) afrag[kc] = load_a_frag_f32(arow, kc, half);

#pragma unroll
    for (int bt = 0; bt < 4; ++bt) {
        v8f acc;
#pragma unroll
        for (int v = 0; v < 8; ++v) acc[v] = bias[v];

        const int n = bt * 16 + ln;  // batch column this lane holds (B-fragment)
        const float* xb = seq + ((size_t)t * BATCH + n) * IDIM;
#pragma unroll
        for (int kc = 0; kc < 4; ++kc) {
            // B layout: lane column n, elements = 16 consecutive k at kc*32+16*half
            const float* p = xb + kc * 32 + 16 * half;
            v4f p0 = *(const v4f*)(p);
            v4f p1 = *(const v4f*)(p + 4);
            v4f p2 = *(const v4f*)(p + 8);
            v4f p3 = *(const v4f*)(p + 12);
            v16bf b;
#pragma unroll
            for (int i = 0; i < 4; ++i) {
                b[i]      = (__bf16)p0[i];
                b[4 + i]  = (__bf16)p1[i];
                b[8 + i]  = (__bf16)p2[i];
                b[12 + i] = (__bf16)p3[i];
            }
            acc = wmma_bf16(afrag[kc], b, acc);
        }
        size_t base = ((((size_t)t_local * 32 + mt) * 4 + bt) * 256) + (size_t)lane * 8;
        *(v8f*)(P + base) = acc;
    }
}

// ---------------------------------------------------------------------------
// Phase 2: persistent recurrent kernel, one workgroup = 16 waves (512 thr).
// Wave w: g = w>>1 (h-rows [g*16,g*16+16)), np = w&1 -> batch tiles {2np, 2np+1}.
// It computes the i/f/g/o tiles for those rows (m-tiles g, 8+g, 16+g, 24+g),
// so the gate nonlinearity fuses entirely in registers; c stays in VGPRs.
// The two batch tiles are processed sequentially, reusing the same W_hh
// A-fragments (keeps peak VGPRs ~200 with a 4-waves/SIMD cap of ~384).
// ---------------------------------------------------------------------------
__global__ __launch_bounds__(512) void lstm_recur_kernel(
    const float* __restrict__ w_hh, const float* __restrict__ P,
    float* __restrict__ out, float* __restrict__ c_state,
    int t0, int ct, int is_first, int is_last) {
    __shared__ __bf16 h_lds[BATCH * HSTRIDE];

    const int tid  = threadIdx.x;
    const int wave = tid >> 5, lane = tid & 31;
    const int half = lane >> 4, ln = lane & 15;
    const int g = wave >> 1, np = wave & 1;

    // W_hh A-fragments, register resident for the whole chunk:
    // 4 gate m-tiles x 4 k-chunks x 8 VGPRs (bf16).
    v16bf a[4][4];
#pragma unroll
    for (int gi = 0; gi < 4; ++gi) {
        const int mt = gi * 8 + g;
        const float* rowbase = w_hh + (size_t)(mt * 16 + ln) * HDIM;
#pragma unroll
        for (int kc = 0; kc < 4; ++kc) a[gi][kc] = load_a_frag_f32(rowbase, kc, half);
    }

    // Seed h (LDS, bf16 col-major [n][k]) and c (registers).
    if (is_first) {
        for (int i = tid; i < BATCH * HSTRIDE; i += 512) h_lds[i] = (__bf16)0.0f;
    } else {
        const float* hprev = out + ((size_t)(t0 - 1) * BATCH) * HDIM;  // hs[t0-1]
        for (int i = tid; i < BATCH * HDIM; i += 512) {
            int b = i >> 7, j = i & 127;
            h_lds[b * HSTRIDE + j] = (__bf16)hprev[(size_t)b * HDIM + j];
        }
    }
    float cc[2][8];
#pragma unroll
    for (int sub = 0; sub < 2; ++sub) {
        if (is_first) {
#pragma unroll
            for (int v = 0; v < 8; ++v) cc[sub][v] = 0.0f;
        } else {
            const int n = (np * 2 + sub) * 16 + ln;
#pragma unroll
            for (int v = 0; v < 8; ++v) {
                int row = g * 16 + v + 8 * half;
                cc[sub][v] = c_state[row * BATCH + n];
            }
        }
    }
    __syncthreads();

    for (int tl = 0; tl < ct; ++tl) {
        const int t = t0 + tl;
        float hv[2][8];

#pragma unroll
        for (int sub = 0; sub < 2; ++sub) {
            const int nt = np * 2 + sub;
            const int n  = nt * 16 + ln;  // batch column this lane owns

            // C-init = precomputed input projection fragments
            v8f acc[4];
#pragma unroll
            for (int gi = 0; gi < 4; ++gi) {
                const int mt = gi * 8 + g;
                size_t base =
                    ((((size_t)tl * 32 + mt) * 4 + nt) * 256) + (size_t)lane * 8;
                acc[gi] = *(const v8f*)(P + base);
            }

            // ifgo += W_hh @ h  (B-frag: 16 contiguous bf16 of column n per lane)
#pragma unroll
            for (int kc = 0; kc < 4; ++kc) {
                const __bf16* p = &h_lds[n * HSTRIDE + kc * 32 + 16 * half];
                v8bf lo = *(const v8bf*)(p);
                v8bf hi = *(const v8bf*)(p + 8);
                v16bf b = __builtin_shufflevector(lo, hi, 0, 1, 2, 3, 4, 5, 6, 7, 8,
                                                  9, 10, 11, 12, 13, 14, 15);
                acc[0] = wmma_bf16(a[0][kc], b, acc[0]);
                acc[1] = wmma_bf16(a[1][kc], b, acc[1]);
                acc[2] = wmma_bf16(a[2][kc], b, acc[2]);
                acc[3] = wmma_bf16(a[3][kc], b, acc[3]);
            }

            // Gate fusion in registers (elem v -> h-row g*16+v+8*half, batch n)
#pragma unroll
            for (int v = 0; v < 8; ++v) {
                float iv = fast_sig(acc[0][v]);
                float fv = fast_sig(acc[1][v]);
                float gv = fast_tanh(acc[2][v]);
                float ov = fast_sig(acc[3][v]);
                float cn = fv * cc[sub][v] + iv * gv;
                cc[sub][v] = cn;
                hv[sub][v] = ov * fast_tanh(cn);
            }
        }

        __syncthreads();  // all waves done reading old h before overwrite

#pragma unroll
        for (int sub = 0; sub < 2; ++sub) {
            const int n = (np * 2 + sub) * 16 + ln;
            {   // h -> LDS (8 consecutive bf16 rows per lane = one 16B store)
                __bf16* q = &h_lds[n * HSTRIDE + g * 16 + 8 * half];
                v8bf hb;
#pragma unroll
                for (int v = 0; v < 8; ++v) hb[v] = (__bf16)hv[sub][v];
                *(v8bf*)q = hb;
            }
            {   // hs[t][b][j] output (fp32)
                float* o = out + ((size_t)t * BATCH + n) * HDIM + g * 16 + 8 * half;
                v4f o0, o1;
#pragma unroll
                for (int v = 0; v < 4; ++v) { o0[v] = hv[sub][v]; o1[v] = hv[sub][v + 4]; }
                *(v4f*)o       = o0;
                *(v4f*)(o + 4) = o1;
                if (is_last && t == T_TOTAL - 1) {
                    size_t off = (size_t)n * HDIM + g * 16 + 8 * half;
                    float* hn = out + (size_t)T_TOTAL * BATCH * HDIM + off;
                    float* cn = hn + (size_t)BATCH * HDIM;
                    v4f c0, c1;
#pragma unroll
                    for (int v = 0; v < 4; ++v) { c0[v] = cc[sub][v]; c1[v] = cc[sub][v + 4]; }
                    *(v4f*)hn       = o0;
                    *(v4f*)(hn + 4) = o1;
                    *(v4f*)cn       = c0;
                    *(v4f*)(cn + 4) = c1;
                }
            }
        }
        __syncthreads();  // new h visible before next step's reads
    }

    // persist c for the next chunk
#pragma unroll
    for (int sub = 0; sub < 2; ++sub) {
        const int n = (np * 2 + sub) * 16 + ln;
#pragma unroll
        for (int v = 0; v < 8; ++v) {
            int row = g * 16 + v + 8 * half;
            c_state[row * BATCH + n] = cc[sub][v];
        }
    }
}

extern "C" void kernel_launch(void* const* d_in, const int* in_sizes, int n_in,
                              void* d_out, int out_size, void* d_ws, size_t ws_size,
                              hipStream_t stream) {
    (void)in_sizes; (void)n_in; (void)out_size;
    const float* seq  = (const float*)d_in[0];  // [T,B,I]
    const float* w_ih = (const float*)d_in[1];  // [4H,I]
    const float* w_hh = (const float*)d_in[2];  // [4H,H]
    const float* b_ih = (const float*)d_in[3];  // [4H,1]
    const float* b_hh = (const float*)d_in[4];  // [4H,1]
    float* out = (float*)d_out;
    float* ws  = (float*)d_ws;

    // Largest time-chunk whose fragment buffer fits the workspace.
    // P chunk = CT * 32 mtiles * 4 ntiles * 256 floats = CT * 128 KB; +64KB c-state.
    int CT = 16;
    const int cands[] = {1024, 512, 256, 128, 64, 32, 16};
    for (int i = 0; i < 7; ++i) {
        size_t need = (size_t)cands[i] * 32768u * 4u + 65536u;
        if (ws_size >= need) { CT = cands[i]; break; }
    }
    float* P       = ws;
    float* c_state = ws + (size_t)CT * 32768u;

    for (int t0 = 0; t0 < T_TOTAL; t0 += CT) {
        lstm_gates_kernel<<<CT * 4, 256, 0, stream>>>(seq, w_ih, b_ih, b_hh, P, t0);
        lstm_recur_kernel<<<1, 512, 0, stream>>>(w_hh, P, out, c_state, t0, CT,
                                                 (t0 == 0) ? 1 : 0,
                                                 (t0 + CT == T_TOTAL) ? 1 : 0);
    }
}
